// GNN_node_Virtualnode_53558242181887
// MI455X (gfx1250) — compile-verified, matrix operand order checked by reading, not run on hardware
//
#include <hip/hip_runtime.h>
#include <hip/hip_bf16.h>

typedef __attribute__((ext_vector_type(16))) __bf16        v16bf;
typedef __attribute__((ext_vector_type(8)))  float         v8f;
typedef __attribute__((ext_vector_type(8)))  unsigned int  v8u;

union FragU { v8u u; v16bf h; };

__device__ __forceinline__ unsigned short f2bf(float x) {
    union { float f; unsigned u; } c; c.f = x;
    unsigned r = c.u + 0x7FFFu + ((c.u >> 16) & 1u);   // round-to-nearest-even
    return (unsigned short)(r >> 16);
}

// fragment stored contiguously (32B per lane): shorts [0..15]
__device__ __forceinline__ v16bf load_frag(const unsigned short* p) {
    const uint4* q = (const uint4*)p;
    uint4 a = q[0], b = q[1];
    FragU f;
    f.u[0] = a.x; f.u[1] = a.y; f.u[2] = a.z; f.u[3] = a.w;
    f.u[4] = b.x; f.u[5] = b.y; f.u[6] = b.z; f.u[7] = b.w;
    return f.h;
}

// fragment from natural row layout: shorts [0..7] and [16..23] (both 16B aligned)
__device__ __forceinline__ v16bf load_frag_split(const unsigned short* p) {
    uint4 a = *(const uint4*)p;
    uint4 b = *(const uint4*)(p + 16);
    FragU f;
    f.u[0] = a.x; f.u[1] = a.y; f.u[2] = a.z; f.u[3] = a.w;
    f.u[4] = b.x; f.u[5] = b.y; f.u[6] = b.z; f.u[7] = b.w;
    return f.h;
}

// ---------------- weight packing / BN folding (prep, cheap) ----------------

// W1: [L][128][256] f32 -> per layer [kc:4][nt:16][lane:32][s:16] bf16
__global__ void pack_w1_kernel(const float* __restrict__ W1, unsigned short* __restrict__ w1p) {
    int tid = blockIdx.x * blockDim.x + threadIdx.x;
    if (tid >= 3 * 32768) return;
    int s    = tid & 15;
    int lane = (tid >> 4) & 31;
    int nt   = (tid >> 9) & 15;
    int kc   = (tid >> 13) & 3;
    int l    = tid >> 15;
    int k = kc * 32 + ((lane < 16) ? 0 : 8) + ((s < 8) ? s : s + 8);
    int n = nt * 16 + (lane & 15);
    w1p[tid] = f2bf(W1[(l * 128 + k) * 256 + n]);
}

// W2: [L][256][128] f32 -> per layer [kc:8][nt:8][lane:32][s:16] bf16
__global__ void pack_w2_kernel(const float* __restrict__ W2, unsigned short* __restrict__ w2p) {
    int tid = blockIdx.x * blockDim.x + threadIdx.x;
    if (tid >= 3 * 32768) return;
    int s    = tid & 15;
    int lane = (tid >> 4) & 31;
    int nt   = (tid >> 9) & 7;
    int kc   = (tid >> 12) & 7;
    int l    = tid >> 15;
    int k = kc * 32 + ((lane < 16) ? 0 : 8) + ((s < 8) ? s : s + 8);
    int n = nt * 16 + (lane & 15);
    w2p[tid] = f2bf(W2[(l * 256 + k) * 128 + n]);
}

// fold bias + BN into per-channel scale/shift
__global__ void fold_bn_kernel(const float* __restrict__ b1,
                               const float* __restrict__ g1, const float* __restrict__ bb1,
                               const float* __restrict__ m1, const float* __restrict__ v1,
                               const float* __restrict__ b2,
                               const float* __restrict__ g2, const float* __restrict__ bb2,
                               const float* __restrict__ m2, const float* __restrict__ v2,
                               float* __restrict__ sc1, float* __restrict__ sh1,
                               float* __restrict__ sc2, float* __restrict__ sh2) {
    int tid = blockIdx.x * blockDim.x + threadIdx.x;
    if (tid < 3 * 256) {
        float s = g1[tid] * rsqrtf(v1[tid] + 1e-5f);
        sc1[tid] = s;
        sh1[tid] = (b1[tid] - m1[tid]) * s + bb1[tid];
    } else if (tid < 3 * 256 + 3 * 128) {
        int i = tid - 3 * 256;
        float s = g2[i] * rsqrtf(v2[i] + 1e-5f);
        sc2[i] = s;
        sh2[i] = (b2[i] - m2[i]) * s + bb2[i];
    }
}

// ---------------- graph kernels ----------------

__global__ void atom_enc_kernel(const int* __restrict__ xa, const float* __restrict__ aemb,
                                float* __restrict__ h, int N) {
    int gid = blockIdx.x * blockDim.x + threadIdx.x;
    int n = gid >> 5;
    if (n >= N) return;
    int c = (gid & 31) * 4;
    float a0 = 0.f, a1 = 0.f, a2 = 0.f, a3 = 0.f;
    #pragma unroll
    for (int f = 0; f < 9; ++f) {
        int idx = xa[n * 9 + f];
        const float4 e = *(const float4*)(aemb + (f * 100 + idx) * 128 + c);
        a0 += e.x; a1 += e.y; a2 += e.z; a3 += e.w;
    }
    float4* o = (float4*)(h + n * 128 + c);
    *o = make_float4(a0, a1, a2, a3);
}

__global__ void vn_init_kernel(float* __restrict__ vn, const float* __restrict__ vn0, int n) {
    int i = blockIdx.x * blockDim.x + threadIdx.x;
    if (i < n) vn[i] = vn0[i & 127];
}

__global__ void zero_kernel(float* __restrict__ p, int n) {
    int i = blockIdx.x * blockDim.x + threadIdx.x;
    if (i < n) p[i] = 0.0f;
}

// h += vn[batch]; agg = 0; vnsum[batch] += h (post-add)
__global__ void node_pre_kernel(float* __restrict__ h, float* __restrict__ agg,
                                const int* __restrict__ batch, const float* __restrict__ vn,
                                float* __restrict__ vnsum, int N, int accum) {
    int gid = blockIdx.x * blockDim.x + threadIdx.x;
    int n = gid >> 5;
    if (n >= N) return;
    int c = (gid & 31) * 4;
    int g = batch[n];
    float4 hv = *(const float4*)(h + n * 128 + c);
    const float4 vv = *(const float4*)(vn + g * 128 + c);
    hv.x += vv.x; hv.y += vv.y; hv.z += vv.z; hv.w += vv.w;
    *(float4*)(h + n * 128 + c) = hv;
    *(float4*)(agg + n * 128 + c) = make_float4(0.f, 0.f, 0.f, 0.f);
    if (accum) {
        float* vs = vnsum + g * 128 + c;
        atomicAdd(vs + 0, hv.x);
        atomicAdd(vs + 1, hv.y);
        atomicAdd(vs + 2, hv.z);
        atomicAdd(vs + 3, hv.w);
    }
}

// wave per edge, lane = 4 channels: msg = relu(h[src] + bond_emb) -> atomicAdd agg[dst]
__global__ void edge_msg_kernel(const float* __restrict__ h, const int* __restrict__ ei,
                                const int* __restrict__ ea, const float* __restrict__ bemb,
                                float* __restrict__ agg, int E) {
    int gid = blockIdx.x * blockDim.x + threadIdx.x;
    int e = gid >> 5;
    if (e >= E) return;
    int c = (gid & 31) * 4;
    int src = ei[e];
    int dst = ei[E + e];
    const float4 hv = *(const float4*)(h + src * 128 + c);
    float x0 = hv.x, x1 = hv.y, x2 = hv.z, x3 = hv.w;
    #pragma unroll
    for (int f = 0; f < 3; ++f) {
        int a = ea[e * 3 + f];
        const float4 b = *(const float4*)(bemb + (f * 10 + a) * 128 + c);
        x0 += b.x; x1 += b.y; x2 += b.z; x3 += b.w;
    }
    x0 = fmaxf(x0, 0.f); x1 = fmaxf(x1, 0.f); x2 = fmaxf(x2, 0.f); x3 = fmaxf(x3, 0.f);
    float* d = agg + dst * 128 + c;
    atomicAdd(d + 0, x0);
    atomicAdd(d + 1, x1);
    atomicAdd(d + 2, x2);
    atomicAdd(d + 3, x3);
}

// ---------------- fused GIN MLP: z=(1+eps)h+agg -> (W1,BN,relu) -> (W2,BN[,relu]) ----------------
// 2 waves/block, 32 rows/wave (two 16-row groups sharing each B fragment -> two
// independent WMMA chains per load, half the L2 weight traffic per row).

#define TSTR 264  // 256 + 8 shorts pad (row stride 528B = 33*16B: aligned & bank-spread)

__global__ __launch_bounds__(64) void gin_mlp_kernel(
    const float* __restrict__ h, const float* __restrict__ agg,
    const float* __restrict__ epsp, int l,
    const unsigned short* __restrict__ w1p, const unsigned short* __restrict__ w2p,
    const float* __restrict__ sc1, const float* __restrict__ sh1,
    const float* __restrict__ sc2, const float* __restrict__ sh2,
    float* __restrict__ hout, int N, int relu_out)
{
    __shared__ unsigned short tls[64 * TSTR];   // hidden activations, bf16

    const int tid    = threadIdx.x;
    const int lane   = tid & 31;
    const int wv     = tid >> 5;                // wave 0..1, owns 32 rows
    const int half8  = (lane < 16) ? 0 : 8;
    const int mloc   = lane & 15;

    const long rb    = (long)blockIdx.x * 64 + wv * 32;   // wave's first row
    const float epsv = 1.0f + epsp[l];

    // ---- build A fragments (z in bf16) for both 16-row groups ----
    v16bf afr[2][4];
    #pragma unroll
    for (int g = 0; g < 2; ++g) {
        long row = rb + g * 16 + mloc;
        if (row > (long)N - 1) row = (long)N - 1;   // clamp (stores are guarded)
        const float* hr = h + row * 128;
        const float* ar = agg + row * 128;
        #pragma unroll
        for (int kc = 0; kc < 4; ++kc) {
            const int k0 = kc * 32 + half8;
            FragU f;
            #pragma unroll
            for (int j = 0; j < 4; ++j) {
                float z0 = fmaf(epsv, hr[k0 + 2 * j],          ar[k0 + 2 * j]);
                float z1 = fmaf(epsv, hr[k0 + 2 * j + 1],      ar[k0 + 2 * j + 1]);
                f.u[j] = (unsigned)f2bf(z0) | ((unsigned)f2bf(z1) << 16);
                float z2 = fmaf(epsv, hr[k0 + 16 + 2 * j],     ar[k0 + 16 + 2 * j]);
                float z3 = fmaf(epsv, hr[k0 + 16 + 2 * j + 1], ar[k0 + 16 + 2 * j + 1]);
                f.u[4 + j] = (unsigned)f2bf(z2) | ((unsigned)f2bf(z3) << 16);
            }
            afr[g][kc] = f.h;
        }
    }

    // ---- stage 1: t = relu(BN(z @ W1 + b1)), 16 output tiles of 16 ----
    for (int nt = 0; nt < 16; ++nt) {
        v8f acc0 = {}, acc1 = {};
        #pragma unroll
        for (int kc = 0; kc < 4; ++kc) {
            v16bf bf = load_frag(w1p + (((kc * 16 + nt) * 32 + lane) << 4));
            acc0 = __builtin_amdgcn_wmma_f32_16x16x32_bf16(false, afr[0][kc], false, bf,
                                                           (short)0, acc0, false, false);
            acc1 = __builtin_amdgcn_wmma_f32_16x16x32_bf16(false, afr[1][kc], false, bf,
                                                           (short)0, acc1, false, false);
        }
        const int n = nt * 16 + mloc;
        const float s = sc1[n], o = sh1[n];
        #pragma unroll
        for (int r = 0; r < 8; ++r) {
            float v0 = fmaxf(fmaf(acc0[r], s, o), 0.0f);
            float v1 = fmaxf(fmaf(acc1[r], s, o), 0.0f);
            tls[(wv * 32 + r + half8) * TSTR + n]      = f2bf(v0);
            tls[(wv * 32 + 16 + r + half8) * TSTR + n] = f2bf(v1);
        }
    }
    __syncthreads();

    // ---- stage 2: out = BN(t @ W2 + b2), K = 256 = 8 chunks of 32 ----
    v16bf tfr[2][8];
    #pragma unroll
    for (int g = 0; g < 2; ++g) {
        const int rl = wv * 32 + g * 16 + mloc;
        #pragma unroll
        for (int kc = 0; kc < 8; ++kc)
            tfr[g][kc] = load_frag_split(&tls[rl * TSTR + kc * 32 + half8]);
    }

    for (int nt = 0; nt < 8; ++nt) {
        v8f acc0 = {}, acc1 = {};
        #pragma unroll
        for (int kc = 0; kc < 8; ++kc) {
            v16bf bf = load_frag(w2p + (((kc * 8 + nt) * 32 + lane) << 4));
            acc0 = __builtin_amdgcn_wmma_f32_16x16x32_bf16(false, tfr[0][kc], false, bf,
                                                           (short)0, acc0, false, false);
            acc1 = __builtin_amdgcn_wmma_f32_16x16x32_bf16(false, tfr[1][kc], false, bf,
                                                           (short)0, acc1, false, false);
        }
        const int n = nt * 16 + mloc;
        const float s = sc2[n], o = sh2[n];
        #pragma unroll
        for (int r = 0; r < 8; ++r) {
            float v0 = fmaf(acc0[r], s, o);
            float v1 = fmaf(acc1[r], s, o);
            if (relu_out) { v0 = fmaxf(v0, 0.0f); v1 = fmaxf(v1, 0.0f); }
            long r0 = rb + r + half8;
            long r1 = rb + 16 + r + half8;
            if (r0 < N) hout[r0 * 128 + n] = v0;
            if (r1 < N) hout[r1 * 128 + n] = v1;
        }
    }
}

// ---------------- virtual-node MLP (G=1024, tiny): block per graph ----------------

__global__ __launch_bounds__(256) void vn_update_kernel(
    const float* __restrict__ vnsum, float* __restrict__ vn,
    const float* __restrict__ W1, const float* __restrict__ b1,
    const float* __restrict__ g1, const float* __restrict__ bb1,
    const float* __restrict__ m1, const float* __restrict__ v1,
    const float* __restrict__ W2, const float* __restrict__ b2,
    const float* __restrict__ g2, const float* __restrict__ bb2,
    const float* __restrict__ m2, const float* __restrict__ v2)
{
    __shared__ float vt[128];
    __shared__ float tb[256];
    const int g = blockIdx.x;
    const int j = threadIdx.x;
    if (j < 128) vt[j] = vnsum[g * 128 + j] + vn[g * 128 + j];  // VP = 1
    __syncthreads();
    float acc = b1[j];
    #pragma unroll 4
    for (int k = 0; k < 128; ++k) acc = fmaf(vt[k], W1[k * 256 + j], acc);
    acc = (acc - m1[j]) * rsqrtf(v1[j] + 1e-5f) * g1[j] + bb1[j];
    tb[j] = fmaxf(acc, 0.0f);
    __syncthreads();
    if (j < 128) {
        float a2 = b2[j];
        #pragma unroll 4
        for (int k = 0; k < 256; ++k) a2 = fmaf(tb[k], W2[k * 128 + j], a2);
        a2 = (a2 - m2[j]) * rsqrtf(v2[j] + 1e-5f) * g2[j] + bb2[j];
        vn[g * 128 + j] = fmaxf(a2, 0.0f);
    }
}

// ---------------- host orchestration ----------------

extern "C" void kernel_launch(void* const* d_in, const int* in_sizes, int n_in,
                              void* d_out, int out_size, void* d_ws, size_t ws_size,
                              hipStream_t stream)
{
    const int N = 50000, E = 800000, G = 1024, L = 3;

    const int*   x_atom     = (const int*)d_in[0];
    const int*   edge_index = (const int*)d_in[1];
    const int*   edge_attr  = (const int*)d_in[2];
    const int*   batch      = (const int*)d_in[3];
    const float* atom_emb   = (const float*)d_in[4];
    const float* vn_emb0    = (const float*)d_in[5];
    const float* bond_emb   = (const float*)d_in[6];
    const float* eps        = (const float*)d_in[7];
    const float* mlp_W1     = (const float*)d_in[8];
    const float* mlp_b1     = (const float*)d_in[9];
    const float* bn1_g      = (const float*)d_in[10];
    const float* bn1_b      = (const float*)d_in[11];
    const float* bn1_m      = (const float*)d_in[12];
    const float* bn1_v      = (const float*)d_in[13];
    const float* mlp_W2     = (const float*)d_in[14];
    const float* mlp_b2     = (const float*)d_in[15];
    const float* bn_g       = (const float*)d_in[16];
    const float* bn_b       = (const float*)d_in[17];
    const float* bn_m       = (const float*)d_in[18];
    const float* bn_v       = (const float*)d_in[19];
    const float* vn_W1      = (const float*)d_in[20];
    const float* vn_b1      = (const float*)d_in[21];
    const float* vn_bn1_g   = (const float*)d_in[22];
    const float* vn_bn1_b   = (const float*)d_in[23];
    const float* vn_bn1_m   = (const float*)d_in[24];
    const float* vn_bn1_v   = (const float*)d_in[25];
    const float* vn_W2      = (const float*)d_in[26];
    const float* vn_b2      = (const float*)d_in[27];
    const float* vn_bn2_g   = (const float*)d_in[28];
    const float* vn_bn2_b   = (const float*)d_in[29];
    const float* vn_bn2_m   = (const float*)d_in[30];
    const float* vn_bn2_v   = (const float*)d_in[31];

    char* ws = (char*)d_ws;
    float*          hA    = (float*)(ws);                               // 25,600,000 B
    float*          agg   = (float*)(ws + 25600000);                    // 25,600,000 B
    float*          vn    = (float*)(ws + 51200000);                    //    524,288 B
    float*          vnsum = (float*)(ws + 51724288);                    //    524,288 B
    unsigned short* w1p   = (unsigned short*)(ws + 52248576);           //    196,608 B
    unsigned short* w2p   = (unsigned short*)(ws + 52445184);           //    196,608 B
    float*          sc1   = (float*)(ws + 52641792);
    float*          sh1   = (float*)(ws + 52644864);
    float*          sc2   = (float*)(ws + 52647936);
    float*          sh2   = (float*)(ws + 52649472);

    // prep: pack weights to bf16 fragments, fold bias+BN
    pack_w1_kernel<<<(3 * 32768 + 255) / 256, 256, 0, stream>>>(mlp_W1, w1p);
    pack_w2_kernel<<<(3 * 32768 + 255) / 256, 256, 0, stream>>>(mlp_W2, w2p);
    fold_bn_kernel<<<(3 * 256 + 3 * 128 + 255) / 256, 256, 0, stream>>>(
        mlp_b1, bn1_g, bn1_b, bn1_m, bn1_v,
        mlp_b2, bn_g, bn_b, bn_m, bn_v, sc1, sh1, sc2, sh2);

    atom_enc_kernel<<<(N * 32 + 255) / 256, 256, 0, stream>>>(x_atom, atom_emb, hA, N);
    vn_init_kernel<<<(G * 128 + 255) / 256, 256, 0, stream>>>(vn, vn_emb0, G * 128);

    float* hin = hA;
    for (int l = 0; l < L; ++l) {
        const int accum = (l < L - 1);
        if (accum) zero_kernel<<<(G * 128 + 255) / 256, 256, 0, stream>>>(vnsum, G * 128);
        node_pre_kernel<<<(N * 32 + 255) / 256, 256, 0, stream>>>(hin, agg, batch, vn, vnsum, N, accum);
        edge_msg_kernel<<<(E * 32 + 255) / 256, 256, 0, stream>>>(
            hin, edge_index, edge_attr, bond_emb + (size_t)l * 3 * 10 * 128, agg, E);

        float* hout = (l == 1) ? hA : (float*)d_out;
        gin_mlp_kernel<<<(N + 63) / 64, 64, 0, stream>>>(
            hin, agg, eps, l,
            w1p + (size_t)l * 32768, w2p + (size_t)l * 32768,
            sc1 + l * 256, sh1 + l * 256, sc2 + l * 128, sh2 + l * 128,
            hout, N, accum);

        if (accum) {
            vn_update_kernel<<<G, 256, 0, stream>>>(
                vnsum, vn,
                vn_W1 + (size_t)l * 128 * 256, vn_b1 + l * 256,
                vn_bn1_g + l * 256, vn_bn1_b + l * 256, vn_bn1_m + l * 256, vn_bn1_v + l * 256,
                vn_W2 + (size_t)l * 256 * 128, vn_b2 + l * 128,
                vn_bn2_g + l * 128, vn_bn2_b + l * 128, vn_bn2_m + l * 128, vn_bn2_v + l * 128);
        }
        hin = hout;
    }
    (void)in_sizes; (void)n_in; (void)out_size; (void)ws_size; (void)hin;
}
